// SummarizeGAT_60258391163617
// MI455X (gfx1250) — matrix operand order repaired and checked by reading, not compiled, not run on hardware
//
#include <hip/hip_runtime.h>

#define NNODES 50000
#define NEDGES 800000
#define ETOT   (NEDGES + NNODES)   // 850000 incl. self loops
#define HEADS  8
#define CDIM   16
#define CHW    128                 // HEADS*CDIM
#define NCLASS 40
#define NLAYER 3
#define NEG_SLOPE 0.2f

typedef __attribute__((ext_vector_type(16))) __bf16 v16bf;
typedef __attribute__((ext_vector_type(8)))  float  v8f;

// ---- order-preserving float<->uint encoding for atomicMax on signed floats ----
__device__ __forceinline__ unsigned fenc(float f) {
  unsigned u = __float_as_uint(f);
  return (u & 0x80000000u) ? ~u : (u | 0x80000000u);
}
__device__ __forceinline__ float fdec(unsigned e) {
  unsigned u = (e & 0x80000000u) ? (e & 0x7FFFFFFFu) : ~e;
  return __uint_as_float(u);
}

// =====================================================================
// WMMA GEMM: C[M,Nc] = A[M,128] * B[128,Nc] (+bias), A/B f32 in memory,
// bf16 WMMA compute, f32 accumulate. grid.x = M/16 (M % 16 == 0).
// blockDim.x = 32 * ceil(Nc/16); each wave computes one 16x16 tile.
// Both A (16x128) and B (128x ncPad, transposed) are staged in LDS as
// bf16 so fragment builds are contiguous ds_load_b128s with no exec
// masking anywhere near the WMMAs.
// Dynamic LDS bytes: (16*128 + ncPad*128) * 2.
// =====================================================================
__global__ void wmma_gemm128(const float* __restrict__ A, const float* __restrict__ B,
                             const float* __restrict__ bias, float* __restrict__ C,
                             int Nc) {
  extern __shared__ __bf16 ldsbuf[];
  __bf16* ldsA  = ldsbuf;             // [16][128]
  __bf16* ldsBt = ldsbuf + 16 * 128;  // [ncPad][128]  (transposed: col-major rows)

  const int wid  = threadIdx.x >> 5;
  const int lane = threadIdx.x & 31;
  const int nw   = blockDim.x >> 5;
  const int ncPad = nw * 16;
  const int rowBase = blockIdx.x * 16;

  // cooperative load of the 16x128 A panel, f32 -> bf16
  for (int i = threadIdx.x; i < 16 * 128; i += blockDim.x) {
    ldsA[i] = (__bf16)A[(size_t)(rowBase + (i >> 7)) * 128 + (i & 127)];
  }
  // cooperative load of B (zero-padded to ncPad cols), transposed into LDS
  for (int k = wid; k < 128; k += nw) {
    for (int c = lane; c < ncPad; c += 32) {
      const float bv = (c < Nc) ? B[(size_t)k * Nc + c] : 0.0f;
      ldsBt[c * 128 + k] = (__bf16)bv;
    }
  }
  __syncthreads();

  const int half = lane >> 4;     // K-half selector (ISA 16-bit A/B layouts)
  const int l16  = lane & 15;     // M row (A) / N col (B)
  const int col  = wid * 16 + l16;

  v8f acc = {};
  #pragma unroll
  for (int kk = 0; kk < 4; ++kk) {          // K = 128 = 4 x 32
    const int k0 = kk * 32;
    v16bf a, b;
    // A fragment: 16x32 bf16, lanes 0-15 / 16-31 hold K-halves per ISA table
    #pragma unroll
    for (int e = 0; e < 16; ++e) {
      const int v = e >> 1, pos = e & 1;
      const int kb = (v < 4) ? (v * 2 + half * 8) : ((v - 4) * 2 + 16 + half * 8);
      a[e] = ldsA[l16 * 128 + k0 + kb + pos];
    }
    // B fragment: 32x16 bf16, lane = column, K-halves per ISA table (same as A)
    #pragma unroll
    for (int e = 0; e < 16; ++e) {
      b[e] = ldsBt[col * 128 + k0 + half * 8 + e + ((e >> 3) << 3)];
    }
    acc = __builtin_amdgcn_wmma_f32_16x16x32_bf16(false, a, false, b,
                                                  (short)0, acc, false, false);
  }

  // C/D layout: VGPR r -> row r (lanes 0-15) / row r+8 (lanes 16-31), col = lane%16
  if (col < Nc) {
    const float bb = bias ? bias[col] : 0.0f;
    const int rofs = half * 8;
    #pragma unroll
    for (int r = 0; r < 8; ++r) {
      C[(size_t)(rowBase + r + rofs) * Nc + col] = acc[r] + bb;
    }
  }
}

// ---- per-(node,head) attention logits: s = <h[n,h,:], a[h,:]> ----
__global__ void sdots_kernel(const float* __restrict__ hlin, const float* __restrict__ a_src,
                             const float* __restrict__ a_dst, float* __restrict__ ssrc,
                             float* __restrict__ sdst) {
  int idx = blockIdx.x * blockDim.x + threadIdx.x;
  if (idx >= NNODES * HEADS) return;
  const int n = idx >> 3, h = idx & 7;
  const float* row = hlin + (size_t)n * CHW + h * CDIM;
  float s0 = 0.f, s1 = 0.f;
  #pragma unroll
  for (int c = 0; c < CDIM; ++c) {
    s0 += row[c] * a_src[h * CDIM + c];
    s1 += row[c] * a_dst[h * CDIM + c];
  }
  ssrc[idx] = s0;
  sdst[idx] = s1;
}

// ---- per-layer accumulator / segment-max / denom init ----
__global__ void init_kernel(float* __restrict__ acc, unsigned* __restrict__ emax,
                            float* __restrict__ denom) {
  int idx = blockIdx.x * blockDim.x + threadIdx.x;
  if (idx < NNODES * CHW) acc[idx] = 0.f;
  if (idx < NNODES * HEADS) {
    emax[idx] = 0x007FFFFFu;   // fenc(-inf)
    denom[idx] = 0.f;
  }
}

__device__ __forceinline__ void edge_sd(int e, const int* src, const int* dst,
                                        int& s, int& d) {
  if (e < NEDGES) { s = src[e]; d = dst[e]; }
  else            { s = e - NEDGES; d = s; }    // self loop
}

__global__ void edge_max_kernel(const int* __restrict__ src, const int* __restrict__ dst,
                                const float* __restrict__ ssrc, const float* __restrict__ sdst,
                                unsigned* __restrict__ emax) {
  int idx = blockIdx.x * blockDim.x + threadIdx.x;
  if (idx >= ETOT * HEADS) return;
  const int e = idx >> 3, h = idx & 7;
  int s, d; edge_sd(e, src, dst, s, d);
  float v = ssrc[s * HEADS + h] + sdst[d * HEADS + h];
  v = (v > 0.f) ? v : NEG_SLOPE * v;
  atomicMax(&emax[d * HEADS + h], fenc(v));
}

__global__ void edge_sum_kernel(const int* __restrict__ src, const int* __restrict__ dst,
                                const float* __restrict__ ssrc, const float* __restrict__ sdst,
                                const unsigned* __restrict__ emax, float* __restrict__ denom) {
  int idx = blockIdx.x * blockDim.x + threadIdx.x;
  if (idx >= ETOT * HEADS) return;
  const int e = idx >> 3, h = idx & 7;
  int s, d; edge_sd(e, src, dst, s, d);
  float v = ssrc[s * HEADS + h] + sdst[d * HEADS + h];
  v = (v > 0.f) ? v : NEG_SLOPE * v;
  atomicAdd(&denom[d * HEADS + h], __expf(v - fdec(emax[d * HEADS + h])));
}

// one wave per edge: 4 strips of 32 channels; alpha-weighted scatter-add
__global__ void edge_agg_kernel(const int* __restrict__ src, const int* __restrict__ dst,
                                const float* __restrict__ hlin, const float* __restrict__ ssrc,
                                const float* __restrict__ sdst, const unsigned* __restrict__ emax,
                                const float* __restrict__ denom, float* __restrict__ acc) {
  const int wid  = (blockIdx.x * blockDim.x + threadIdx.x) >> 5;
  const int lane = threadIdx.x & 31;
  if (wid >= ETOT) return;
  int s, d; edge_sd(wid, src, dst, s, d);
  #pragma unroll
  for (int ch = 0; ch < 4; ++ch) {
    const int c = ch * 32 + lane;
    const int h = c >> 4;
    float v = ssrc[s * HEADS + h] + sdst[d * HEADS + h];
    v = (v > 0.f) ? v : NEG_SLOPE * v;
    const float alpha = __expf(v - fdec(emax[d * HEADS + h])) /
                        (denom[d * HEADS + h] + 1e-16f);
    atomicAdd(&acc[(size_t)d * CHW + c], hlin[(size_t)s * CHW + c] * alpha);
  }
}

// bias + ELU
__global__ void finalize_kernel(const float* __restrict__ acc, const float* __restrict__ bias,
                                float* __restrict__ hout) {
  int idx = blockIdx.x * blockDim.x + threadIdx.x;
  if (idx >= NNODES * CHW) return;
  const float v = acc[idx] + bias[idx & 127];
  hout[idx] = (v > 0.f) ? v : (__expf(v) - 1.f);
}

// scores[n,l] = sum_c tanh(g[n,c]) * aatt[c]
__global__ void score_kernel(const float* __restrict__ g, const float* __restrict__ aatt,
                             float* __restrict__ scores, int l) {
  int n = blockIdx.x * blockDim.x + threadIdx.x;
  if (n >= NNODES) return;
  const float* row = g + (size_t)n * CHW;
  float s = 0.f;
  for (int c = 0; c < CHW; ++c) s += tanhf(row[c]) * aatt[c];
  scores[n * NLAYER + l] = s;
}

// in-place softmax over the 3 layer scores
__global__ void lam_kernel(float* __restrict__ scores) {
  int n = blockIdx.x * blockDim.x + threadIdx.x;
  if (n >= NNODES) return;
  float s0 = scores[n * 3 + 0], s1 = scores[n * 3 + 1], s2 = scores[n * 3 + 2];
  const float m = fmaxf(s0, fmaxf(s1, s2));
  const float e0 = __expf(s0 - m), e1 = __expf(s1 - m), e2 = __expf(s2 - m);
  const float inv = 1.f / (e0 + e1 + e2);
  scores[n * 3 + 0] = e0 * inv;
  scores[n * 3 + 1] = e1 * inv;
  scores[n * 3 + 2] = e2 * inv;
}

__global__ void hfinal_kernel(const float* __restrict__ hs0, const float* __restrict__ hs1,
                              const float* __restrict__ hs2, const float* __restrict__ lam,
                              float* __restrict__ hf) {
  int idx = blockIdx.x * blockDim.x + threadIdx.x;
  if (idx >= NNODES * CHW) return;
  const int n = idx >> 7;
  hf[idx] = lam[n * 3 + 0] * hs0[idx] + lam[n * 3 + 1] * hs1[idx] +
            lam[n * 3 + 2] * hs2[idx];
}

// =====================================================================
extern "C" void kernel_launch(void* const* d_in, const int* in_sizes, int n_in,
                              void* d_out, int out_size, void* d_ws, size_t ws_size,
                              hipStream_t stream) {
  (void)in_sizes; (void)n_in; (void)out_size; (void)ws_size;

  const float* x    = (const float*)d_in[0];
  const int*   eidx = (const int*)  d_in[1];
  const int*   src  = eidx;
  const int*   dst  = eidx + NEDGES;
  const float* W0   = (const float*)d_in[2];
  const float* as0  = (const float*)d_in[3];
  const float* ad0  = (const float*)d_in[4];
  const float* b0   = (const float*)d_in[5];
  const float* W12  = (const float*)d_in[6];
  const float* as12 = (const float*)d_in[7];
  const float* ad12 = (const float*)d_in[8];
  const float* b12  = (const float*)d_in[9];
  const float* Watt = (const float*)d_in[10];
  const float* aatt = (const float*)d_in[11];
  const float* Wout = (const float*)d_in[12];
  const float* bout = (const float*)d_in[13];
  float* out = (float*)d_out;

  // ---- workspace carve-up (floats) ----
  const size_t NCH = (size_t)NNODES * CHW;     // 6.4M
  float* wsf   = (float*)d_ws;
  float* hs0   = wsf;
  float* hs1   = hs0 + NCH;
  float* hs2   = hs1 + NCH;
  float* hlin  = hs2 + NCH;          // later reused as gbuf
  float* accb  = hlin + NCH;         // later reused as hfinal
  float* ssrc  = accb + NCH;
  float* sdst  = ssrc + (size_t)NNODES * HEADS;
  unsigned* emax = (unsigned*)(sdst + (size_t)NNODES * HEADS);
  float* denom = (float*)(emax + (size_t)NNODES * HEADS);
  float* scores = denom + (size_t)NNODES * HEADS;   // N*3

  const int TPB = 256;
  const dim3 gNCH((NNODES * CHW + TPB - 1) / TPB);
  const dim3 gEH((ETOT * HEADS + TPB - 1) / TPB);
  const dim3 gN((NNODES + TPB - 1) / TPB);
  const dim3 gEdgeWaves((ETOT * 32 + TPB - 1) / TPB);   // one wave per edge
  const int MT = NNODES / 16;                           // 3125 row tiles

  // dynamic LDS sizes for the two GEMM shapes (A panel + transposed B panel, bf16)
  const size_t shm128 = (size_t)(16 * 128 + 128 * 128) * sizeof(__bf16);  // 36 KB
  const size_t shm48  = (size_t)(16 * 128 +  48 * 128) * sizeof(__bf16);  // 16 KB

  float* hsArr[NLAYER] = { hs0, hs1, hs2 };
  const float* act = x;

  for (int l = 0; l < NLAYER; ++l) {
    const float* W   = (l == 0) ? W0  : W12  + (size_t)(l - 1) * CHW * CHW;
    const float* as_ = (l == 0) ? as0 : as12 + (size_t)(l - 1) * HEADS * CDIM;
    const float* ad_ = (l == 0) ? ad0 : ad12 + (size_t)(l - 1) * HEADS * CDIM;
    const float* bb  = (l == 0) ? b0  : b12  + (size_t)(l - 1) * CHW;

    wmma_gemm128<<<MT, 256, shm128, stream>>>(act, W, nullptr, hlin, CHW);
    sdots_kernel<<<(NNODES * HEADS + TPB - 1) / TPB, TPB, 0, stream>>>(hlin, as_, ad_, ssrc, sdst);
    init_kernel<<<gNCH, TPB, 0, stream>>>(accb, emax, denom);
    edge_max_kernel<<<gEH, TPB, 0, stream>>>(src, dst, ssrc, sdst, emax);
    edge_sum_kernel<<<gEH, TPB, 0, stream>>>(src, dst, ssrc, sdst, emax, denom);
    edge_agg_kernel<<<gEdgeWaves, TPB, 0, stream>>>(src, dst, hlin, ssrc, sdst, emax, denom, accb);
    finalize_kernel<<<gNCH, TPB, 0, stream>>>(accb, bb, hsArr[l]);
    act = hsArr[l];
  }

  // ---- attention-based skip aggregation over layers ----
  for (int l = 0; l < NLAYER; ++l) {
    wmma_gemm128<<<MT, 256, shm128, stream>>>(hsArr[l], Watt, nullptr, hlin /*gbuf*/, CHW);
    score_kernel<<<gN, TPB, 0, stream>>>(hlin, aatt, scores, l);
  }
  lam_kernel<<<gN, TPB, 0, stream>>>(scores);
  hfinal_kernel<<<gNCH, TPB, 0, stream>>>(hs0, hs1, hs2, scores, accb /*hfinal*/);

  // ---- output head: [N,128] x [128,40] + bout, 3 column tiles (pad to 48) ----
  wmma_gemm128<<<MT, 96, shm48, stream>>>(accb, Wout, bout, out, NCLASS);
}
